// Local_Scale_Attention_82454782149134
// MI455X (gfx1250) — compile-verified, matrix operand order checked by reading, not run on hardware
//
#include <hip/hip_runtime.h>

typedef __attribute__((ext_vector_type(16))) __bf16 v16bf;
typedef __attribute__((ext_vector_type(8)))  float  v8f;

#define N_TOK 84
#define NPAD 96
#define DIM 256
#define HEADS 8
#define HD 32
#define TABLE_ROWS 207

__device__ __forceinline__ unsigned short f2bf(float f) {
  unsigned u = __float_as_uint(f);
  u = (u + 0x7FFFu + ((u >> 16) & 1u)) >> 16;   // round-to-nearest-even
  return (unsigned short)u;
}

// Load a 16x32 bf16 WMMA fragment (A row-major, or B from a transposed-stored
// matrix). Per-lane layout (CDNA5 ISA 7.12.2): lane l (r=l&15, half=l>>4) holds
// row r with K offsets [half*8, half*8+8) and [16+half*8, 16+half*8+8).
__device__ __forceinline__ v16bf frag16(const unsigned short* p, int row0,
                                        int stride, int kbase, int lane) {
  const unsigned short* base =
      p + (row0 + (lane & 15)) * stride + kbase + ((lane >> 4) << 3);
  union { v16bf v; uint4 q[2]; } u;
  u.q[0] = *(const uint4*)(base);
  u.q[1] = *(const uint4*)(base + 16);
  return u.v;
}

// Convert W_qkv (256x768) and W_proj (256x256) f32 -> bf16, transposed so
// B-fragment loads are contiguous (row = output column, 256 K-elements each).
__global__ void prep_weights(const float* __restrict__ Wqkv,
                             const float* __restrict__ Wproj,
                             unsigned short* __restrict__ wsQ,
                             unsigned short* __restrict__ wsP) {
  int i = blockIdx.x * blockDim.x + threadIdx.x;
  const int n1 = 768 * 256;
  const int n2 = 256 * 256;
  if (i < n1) {
    int o = i / 256, k = i % 256;
    wsQ[i] = f2bf(Wqkv[k * 768 + o]);
  } else if (i < n1 + n2) {
    int j = i - n1;
    int o = j / 256, k = j % 256;
    wsP[j] = f2bf(Wproj[k * 256 + o]);
  }
}

__global__ __launch_bounds__(256) void Local_Scale_Attention_kernel(
    const float* __restrict__ x, const float* __restrict__ b_qkv,
    const float* __restrict__ b_proj, const float* __restrict__ bias_table,
    const int* __restrict__ rel_index,
    const unsigned short* __restrict__ wsQ,
    const unsigned short* __restrict__ wsP, float* __restrict__ out) {
  __shared__ __align__(16) unsigned short sXb[NPAD * DIM];   // x[b] bf16     48 KB
  __shared__ __align__(16) unsigned short sO [NPAD * DIM];   // attn out bf16 48 KB
  __shared__ __align__(16) unsigned short sQ [NPAD * HD];    // per-head Q     6 KB
  __shared__ __align__(16) unsigned short sK [NPAD * HD];    // per-head K     6 KB
  __shared__ __align__(16) unsigned short sVt[HD * NPAD];    // per-head V^T   6 KB
  __shared__ __align__(16) unsigned short sP [NPAD * NPAD];  // probs bf16    18 KB
  __shared__ unsigned char sRel[NPAD * NPAD];                //                9 KB
  __shared__ float sTab[TABLE_ROWS * HEADS];                 //              6.6 KB

  const int t = threadIdx.x;
  const int lane = t & 31;
  const int w = t >> 5;
  const int c15 = lane & 15;
  const int half8 = (lane >> 4) << 3;
  const long b = blockIdx.x;
  const float scale = 0.10910894511799618f;  // 84^-0.5

  // ---- Phase A: stage x[b] (bf16, zero-padded), rel_index (u8), bias table
  const float* xb = x + b * (N_TOK * DIM);
  for (int i = t * 2; i < NPAD * DIM; i += 512) {
    int r = i >> 8, c = i & 255;                 // i even -> c,c+1 in same row
    float v0 = 0.f, v1 = 0.f;
    if (r < N_TOK) { v0 = xb[r * DIM + c]; v1 = xb[r * DIM + c + 1]; }
    *(unsigned int*)&sXb[i] = (unsigned)f2bf(v0) | ((unsigned)f2bf(v1) << 16);
  }
  for (int i = t; i < NPAD * NPAD; i += 256) {
    int r = i / NPAD, c = i % NPAD;
    sRel[i] = (r < N_TOK && c < N_TOK) ? (unsigned char)rel_index[r * N_TOK + c] : 0;
  }
  for (int i = t; i < TABLE_ROWS * HEADS; i += 256) sTab[i] = bias_table[i];
  __syncthreads();

  for (int h = 0; h < HEADS; ++h) {
    // ---- Phase B: per-head QKV GEMM. Wave w (<6) owns one 16-wide output
    // column strip (nt = w) and reuses each weight B-fragment across all 6
    // M-tiles (6x less L2 weight traffic).
    if (w < 6) {
      int nt = w, qsel = nt >> 1, sub = (nt & 1) * 16;
      int wcol = qsel * 256 + h * 32 + sub;   // (3,H,hd) column split
      float bias = b_qkv[wcol + c15];
      v8f acc[6] = {};
      for (int kk = 0; kk < 8; ++kk) {
        v16bf bw = frag16(wsQ, wcol, DIM, kk * 32, lane);
        #pragma unroll
        for (int mt = 0; mt < 6; ++mt) {
          v16bf a = frag16(sXb, mt * 16, DIM, kk * 32, lane);
          acc[mt] = __builtin_amdgcn_wmma_f32_16x16x32_bf16(
              false, a, false, bw, (short)0, acc[mt], false, false);
        }
      }
      // branch-free store: pick base/stride once per strip
      unsigned short* basePtr; int strideM, colOff;
      if (qsel == 0)      { basePtr = sQ;  strideM = HD; colOff = sub + c15; }
      else if (qsel == 1) { basePtr = sK;  strideM = HD; colOff = sub + c15; }
      else                { basePtr = sVt; strideM = 1;  colOff = (sub + c15) * NPAD; }
      #pragma unroll
      for (int mt = 0; mt < 6; ++mt)
        #pragma unroll
        for (int g = 0; g < 8; ++g) {
          int m = mt * 16 + half8 + g;
          basePtr[colOff + m * strideM] = f2bf(acc[mt][g] + bias);
        }
    }
    __syncthreads();

    // ---- Phase C1: scores + softmax (waves 0..5, one M-tile each; K=hd=32)
    if (w < 6) {
      int mt = w;
      float s[6][8];
      v16bf aq = frag16(sQ, mt * 16, HD, 0, lane);
      #pragma unroll
      for (int nt = 0; nt < 6; ++nt) {
        v16bf bk = frag16(sK, nt * 16, HD, 0, lane);   // K^T fragment
        v8f acc = {};
        acc = __builtin_amdgcn_wmma_f32_16x16x32_bf16(false, aq, false, bk,
                                                      (short)0, acc, false, false);
        #pragma unroll
        for (int g = 0; g < 8; ++g) {
          int m = mt * 16 + half8 + g;
          int n = nt * 16 + c15;
          float v = acc[g] * scale + sTab[(int)sRel[m * NPAD + n] * HEADS + h];
          s[nt][g] = (n < N_TOK) ? v : -INFINITY;
        }
      }
      float rmax[8], rsum[8];
      #pragma unroll
      for (int g = 0; g < 8; ++g) {
        float m = s[0][g];
        #pragma unroll
        for (int nt = 1; nt < 6; ++nt) m = fmaxf(m, s[nt][g]);
        for (int d = 1; d < 16; d <<= 1) m = fmaxf(m, __shfl_xor(m, d, 32));
        float sum = 0.f;
        #pragma unroll
        for (int nt = 0; nt < 6; ++nt) sum += __expf(s[nt][g] - m);
        for (int d = 1; d < 16; d <<= 1) sum += __shfl_xor(sum, d, 32);
        rmax[g] = m; rsum[g] = sum;
      }
      #pragma unroll
      for (int nt = 0; nt < 6; ++nt)
        #pragma unroll
        for (int g = 0; g < 8; ++g) {
          int m = mt * 16 + half8 + g;
          int n = nt * 16 + c15;
          sP[m * NPAD + n] = f2bf(__expf(s[nt][g] - rmax[g]) / rsum[g]);
        }
    }
    __syncthreads();

    // ---- Phase C2: P @ V (12 tiles, K=96), all LDS operands
    for (int idx = w; idx < 12; idx += 8) {
      int mt = idx / 2, dt = idx % 2;
      v8f acc = {};
      #pragma unroll
      for (int kt = 0; kt < 3; ++kt) {
        v16bf a  = frag16(sP, mt * 16, NPAD, kt * 32, lane);
        v16bf bv = frag16(sVt, dt * 16, NPAD, kt * 32, lane);
        acc = __builtin_amdgcn_wmma_f32_16x16x32_bf16(false, a, false, bv,
                                                      (short)0, acc, false, false);
      }
      #pragma unroll
      for (int g = 0; g < 8; ++g) {
        int m = mt * 16 + half8 + g;
        sO[m * DIM + h * 32 + dt * 16 + c15] = f2bf(acc[g]);
      }
    }
    __syncthreads();
  }

  // ---- Phase D: output projection. Each wave owns 2 column strips and
  // reuses each W_proj B-fragment across all 6 M-tiles.
  float* ob = out + b * (N_TOK * DIM);
  #pragma unroll
  for (int nn = 0; nn < 2; ++nn) {
    int nt = w * 2 + nn;
    float bias = b_proj[nt * 16 + c15];
    v8f acc[6] = {};
    for (int kk = 0; kk < 8; ++kk) {
      v16bf bw = frag16(wsP, nt * 16, DIM, kk * 32, lane);
      #pragma unroll
      for (int mt = 0; mt < 6; ++mt) {
        v16bf a = frag16(sO, mt * 16, DIM, kk * 32, lane);
        acc[mt] = __builtin_amdgcn_wmma_f32_16x16x32_bf16(
            false, a, false, bw, (short)0, acc[mt], false, false);
      }
    }
    #pragma unroll
    for (int mt = 0; mt < 6; ++mt)
      #pragma unroll
      for (int g = 0; g < 8; ++g) {
        int m = mt * 16 + half8 + g;
        if (m < N_TOK) ob[m * DIM + nt * 16 + c15] = acc[mt][g] + bias;
      }
  }
}

extern "C" void kernel_launch(void* const* d_in, const int* in_sizes, int n_in,
                              void* d_out, int out_size, void* d_ws, size_t ws_size,
                              hipStream_t stream) {
  const float* x          = (const float*)d_in[0];
  const float* W_qkv      = (const float*)d_in[1];
  const float* b_qkv      = (const float*)d_in[2];
  const float* W_proj     = (const float*)d_in[3];
  const float* b_proj     = (const float*)d_in[4];
  const float* bias_table = (const float*)d_in[5];
  const int*   rel_index  = (const int*)d_in[6];
  float* out = (float*)d_out;

  unsigned short* wsQ = (unsigned short*)d_ws;       // 768x256 bf16 (W_qkv^T)
  unsigned short* wsP = wsQ + 768 * 256;             // 256x256 bf16 (W_proj^T)

  int B = in_sizes[0] / (N_TOK * DIM);
  int nconv = 768 * 256 + 256 * 256;
  hipLaunchKernelGGL(prep_weights, dim3((nconv + 255) / 256), dim3(256), 0, stream,
                     W_qkv, W_proj, wsQ, wsP);
  hipLaunchKernelGGL(Local_Scale_Attention_kernel, dim3(B), dim3(256), 0, stream,
                     x, b_qkv, b_proj, bias_table, rel_index, wsQ, wsP, out);
  (void)n_in; (void)out_size; (void)ws_size;
}
